// CRFlayer_12644383719425
// MI455X (gfx1250) — matrix operand order repaired and checked by reading, not compiled, not run on hardware
//
#include <hip/hip_runtime.h>
#include <hip/hip_bf16.h>
#include <stdint.h>

// CRF loss on MI455X (gfx1250, wave32, WMMA).
// T=512 sequential scan; per step a (1024x64)x(64x64) GEMM in f16 WMMA with
// E=exp(trans) split into hi+lo f16 parts for ~f32 accuracy.
// Latency-optimized serial chain: ONE barrier per step implemented as raw
// s_wait_dscnt+s_barrier (no loadcnt drain -> global prefetch stays in
// flight); exp/log via raw v_exp_f32/v_log_f32 with fma-folded bases;
// one-step-lagged max reference (+3 shift keeps p in f16 normal range;
// mathematically exact since the reference cancels in cu + log(dot)).

#define T_ 512
#define B_ 1024
#define N_ 64

typedef __attribute__((ext_vector_type(16))) _Float16 v16h;
typedef __attribute__((ext_vector_type(8)))  _Float16 v8h;
typedef __attribute__((ext_vector_type(8)))  float    v8f;

union V16U { v16h v; v8h h8[2]; };

#define LOG2E_ 1.4426950408889634f
#define LN2_   0.6931471805599453f

__device__ inline unsigned pack_pkrtz(float x, float y) {
    auto h = __builtin_amdgcn_cvt_pkrtz(x, y);   // v_cvt_pk_rtz_f16_f32
    union { decltype(h) v; unsigned u; } cv; cv.v = h; return cv.u;
}

// Workgroup barrier that only drains LDS (dscnt), leaving global loads in
// flight across it. (__syncthreads drains loadcnt/storecnt too.)
__device__ inline void lds_barrier() {
    __asm__ volatile("s_wait_dscnt 0\n\t"
                     "s_barrier_signal -1\n\t"
                     "s_barrier_wait -1" ::: "memory");
}

// ---------------------------------------------------------------- prep: E^T hi/lo
__global__ void crf_prep(const float* __restrict__ trans,
                         _Float16* __restrict__ EThi,
                         _Float16* __restrict__ ETlo) {
    int idx = blockIdx.x * blockDim.x + threadIdx.x;   // idx = j*64 + i
    if (idx < N_ * N_) {
        int j = idx >> 6, i = idx & 63;
        float e = __expf(trans[i * N_ + j]);           // transposed store
        _Float16 hi = (_Float16)e;
        EThi[idx] = hi;
        ETlo[idx] = (_Float16)(e - (float)hi);
    }
}

// ---------------------------------------------------------------- forward + fused score
// Grid: 64 WGs x 128 threads. WG owns 16 batches; wave w owns j-tile w.
// alpha tile layout == WMMA C/D: VGPR r, lanes 0-15: j=16w+r, b=lane;
// lanes 16-31: j=16w+8+r, b=lane-16.
__global__ void __launch_bounds__(128)
crf_forward(const float* __restrict__ emit,
            const uint8_t* __restrict__ mask,        // jnp bool -> 1 byte
            const float* __restrict__ strans,
            const float* __restrict__ etrans,
            const float* __restrict__ trans,
            const int* __restrict__ target,
            const _Float16* __restrict__ EThi,
            const _Float16* __restrict__ ETlo,
            float* __restrict__ logZpart,
            float* __restrict__ scpart) {
    __shared__ __align__(16) uint4 Pb[2][128];   // p^T fragments (parity buffered)
    __shared__ __align__(16) float pmb[2][64];   // per-wave partial max, [b][w]
    __shared__ __align__(16) float psum[64];     // bootstrap max / final expsum
    __shared__ float sred[128];

    const int tid  = threadIdx.x;
    const int w    = tid >> 5;           // j-tile 0..3
    const int lane = tid & 31;
    const int h    = lane >> 4;          // lane half
    const int bl   = lane & 15;          // local batch
    const int bb   = blockIdx.x * 16;
    const int b    = bb + bl;
    const int jr   = 16 * w + 8 * h;     // j of register r is jr + r
    const int j0   = 16 * w + bl;        // A-operand row (M = lane%16)

    // Resident A operands: E^T hi/lo, 2 K-tiles (A 16x32 layout:
    // lanes0-15 comps0-7 -> K=0..7, comps8-15 -> K=16..23; lanes16-31 shifted by 8).
    V16U ahi[2], alo[2];
#pragma unroll
    for (int kt = 0; kt < 2; ++kt) {
        int base = j0 * N_ + 32 * kt + 8 * h;
        ahi[kt].h8[0] = *(const v8h*)(EThi + base);
        ahi[kt].h8[1] = *(const v8h*)(EThi + base + 16);
        alo[kt].h8[0] = *(const v8h*)(ETlo + base);
        alo[kt].h8[1] = *(const v8h*)(ETlo + base + 16);
    }

    // t = 0: alpha = emit[0] + strans ; fused score init (branchless)
    float a[8];
    float sacc = 0.f;
    int   tg_prev = target[b];
    int   last = tg_prev;
    {
        const float* e0 = emit + (size_t)b * N_ + jr;
        const float* s0 = strans + jr;
        float m0f = mask[b] ? 1.f : 0.f;
        float pick = 0.f;
#pragma unroll
        for (int r = 0; r < 8; ++r) {
            float er = e0[r];
            a[r] = er + s0[r];
            pick += (tg_prev == jr + r) ? er : 0.f;
        }
        sacc += m0f * pick;
        if (w == 0) sacc += (h == 0) ? strans[tg_prev] : 0.f;  // once per batch
    }

    // prefetch emit/mask/target for t = 1
    float ev[8]; uint8_t mv; int tg;
    {
        const float* ep = emit + ((size_t)B_ + b) * N_ + jr;
#pragma unroll
        for (int r = 0; r < 8; ++r) ev[r] = ep[r];
        mv = mask[B_ + b];
        tg = target[B_ + b];
    }

    // bootstrap: exact per-batch max of alpha0 -> cu (uses psum scratch)
    float cu;
    {
        float pm = a[0];
#pragma unroll
        for (int r = 1; r < 8; ++r) pm = fmaxf(pm, a[r]);
        pm = fmaxf(pm, __shfl_xor(pm, 16));
        if (h == 0) psum[bl * 4 + w] = pm;
        __syncthreads();
        float4 q = *(const float4*)&psum[bl * 4];
        cu = fmaxf(fmaxf(q.x, q.y), fmaxf(q.z, q.w)) + 3.0f;
    }

#pragma unroll 2
    for (int t = 1; t < T_; ++t) {
        const int par = t & 1;

        // prefetch next step early (clamped -> branch-free body); these loads
        // stay outstanding across lds_barrier (no loadcnt drain)
        const int tn = (t + 1 < T_) ? (t + 1) : (T_ - 1);
        float evn[8]; uint8_t mvn; int tgn;
        {
            const float* ep = emit + ((size_t)tn * B_ + b) * N_ + jr;
#pragma unroll
            for (int r = 0; r < 8; ++r) evn[r] = ep[r];
            mvn = mask[tn * B_ + b];
            tgn = target[tn * B_ + b];
        }

        // partial max of current alpha -> pmb[par] (combined AFTER the barrier,
        // consumed as next iteration's exp reference: one-step lag)
        float pm = a[0];
#pragma unroll
        for (int r = 1; r < 8; ++r) pm = fmaxf(pm, a[r]);
        pm = fmaxf(pm, __shfl_xor(pm, 16));
        if (h == 0) pmb[par][bl * 4 + w] = pm;

        // p = exp2((a - cu)*log2e) -> f16 pairs -> one ds_store_b128 per lane
        const float cul = cu * LOG2E_;
        float p[8];
#pragma unroll
        for (int r = 0; r < 8; ++r)
            p[r] = __builtin_amdgcn_exp2f(__builtin_fmaf(a[r], LOG2E_, -cul));
        uint4 pk;
        pk.x = pack_pkrtz(p[0], p[1]);
        pk.y = pack_pkrtz(p[2], p[3]);
        pk.z = pack_pkrtz(p[4], p[5]);
        pk.w = pack_pkrtz(p[6], p[7]);
        Pb[par][(2 * w + h) * 16 + bl] = pk;

        lds_barrier();     // single barrier per step, drains dscnt only

        // combine lagged max (off the WMMA critical path)
        float4 q = *(const float4*)&pmb[par][bl * 4];
        float mNew = fmaxf(fmaxf(q.x, q.y), fmaxf(q.z, q.w));

        // B operand (p^T, K x 16): lanes0-15 K=0..7,16..23, lanes16-31 K=8..15,24..31 per K-tile
        V16U bp0, bp1;
        bp0.h8[0] = *(const v8h*)&Pb[par][(0 + h) * 16 + bl];
        bp0.h8[1] = *(const v8h*)&Pb[par][(2 + h) * 16 + bl];
        bp1.h8[0] = *(const v8h*)&Pb[par][(4 + h) * 16 + bl];
        bp1.h8[1] = *(const v8h*)&Pb[par][(6 + h) * 16 + bl];

        v8f acc = {};
        acc = __builtin_amdgcn_wmma_f32_16x16x32_f16(false, ahi[0].v, false, bp0.v, (short)0, acc, false, false);
        acc = __builtin_amdgcn_wmma_f32_16x16x32_f16(false, alo[0].v, false, bp0.v, (short)0, acc, false, false);
        acc = __builtin_amdgcn_wmma_f32_16x16x32_f16(false, ahi[1].v, false, bp1.v, (short)0, acc, false, false);
        acc = __builtin_amdgcn_wmma_f32_16x16x32_f16(false, alo[1].v, false, bp1.v, (short)0, acc, false, false);

        // fused path score for step t (branchless; one wave-uniform branch)
        {
            float mvf = mv ? 1.f : 0.f;
            float pick = 0.f;
#pragma unroll
            for (int r = 0; r < 8; ++r) pick += (tg == jr + r) ? ev[r] : 0.f;
            sacc += mvf * pick;
            if (w == 0) {                                   // wave-uniform branch
                float tr = trans[tg_prev * N_ + tg];        // unconditional gather
                sacc += (h == 0 && mv) ? tr : 0.f;
            }
            last = mv ? tg : last;
            tg_prev = tg;
        }

        // alpha update: cu + ln2*log2(dot) + emit (masked select, no denorm path)
#pragma unroll
        for (int r = 0; r < 8; ++r) {
            float nv = __builtin_fmaf(__builtin_amdgcn_logf(acc[r]), LN2_, cu + ev[r]);
            a[r] = mv ? nv : a[r];
        }
#pragma unroll
        for (int r = 0; r < 8; ++r) ev[r] = evn[r];
        mv = mvn;
        tg = tgn;
        cu = mNew + 3.0f;   // next step's exp reference (one-step lag)
    }

    // logZ partial: sum_b logsumexp_j(alpha + etrans)
    {
        const float* et = etrans + jr;
#pragma unroll
        for (int r = 0; r < 8; ++r) a[r] += et[r];
        float pm = a[0];
#pragma unroll
        for (int r = 1; r < 8; ++r) pm = fmaxf(pm, a[r]);
        pm = fmaxf(pm, __shfl_xor(pm, 16));
        if (h == 0) pmb[0][bl * 4 + w] = pm;
        __syncthreads();
        float4 q = *(const float4*)&pmb[0][bl * 4];
        float c = fmaxf(fmaxf(q.x, q.y), fmaxf(q.z, q.w));
        float s = 0.f;
#pragma unroll
        for (int r = 0; r < 8; ++r) s += __expf(a[r] - c);
        s += __shfl_xor(s, 16);
        if (h == 0) psum[bl * 4 + w] = s;
        __syncthreads();
        float4 sq = *(const float4*)&psum[bl * 4];
        float lz = c + __logf(sq.x + sq.y + sq.z + sq.w);
        if (w == 0) {
            float tot = lz;                       // lanes 0..15 reduce among themselves
            tot += __shfl_xor(tot, 1);
            tot += __shfl_xor(tot, 2);
            tot += __shfl_xor(tot, 4);
            tot += __shfl_xor(tot, 8);
            if (lane == 0) logZpart[blockIdx.x] = tot;
        }
    }

    // score partial: tree reduce 128 lanes (fixed order -> deterministic)
    if (w == 0) sacc += (h == 0) ? etrans[last] : 0.f;
    sred[tid] = sacc;
    __syncthreads();
#pragma unroll
    for (int off = 64; off > 0; off >>= 1) {
        if (tid < off) sred[tid] += sred[tid + off];
        __syncthreads();
    }
    if (tid == 0) scpart[blockIdx.x] = sred[0];
}

// ---------------------------------------------------------------- final scalar
__global__ void crf_final(const float* __restrict__ logZpart,
                          const float* __restrict__ scpart,
                          float* __restrict__ out) {
    float lz = 0.f, sc = 0.f;
    for (int i = 0; i < 64; ++i) { lz += logZpart[i]; sc += scpart[i]; }
    out[0] = (lz - sc) / (float)B_;
}

// ---------------------------------------------------------------- launch
extern "C" void kernel_launch(void* const* d_in, const int* in_sizes, int n_in,
                              void* d_out, int out_size, void* d_ws, size_t ws_size,
                              hipStream_t stream) {
    (void)in_sizes; (void)n_in; (void)out_size; (void)ws_size;
    const float*   emit   = (const float*)d_in[0];
    const float*   trans  = (const float*)d_in[1];
    const float*   strans = (const float*)d_in[2];
    const float*   etrans = (const float*)d_in[3];
    const int*     target = (const int*)d_in[4];
    const uint8_t* mask   = (const uint8_t*)d_in[5];

    _Float16* EThi = (_Float16*)d_ws;
    _Float16* ETlo = EThi + N_ * N_;
    float* logZpart = (float*)(ETlo + N_ * N_);
    float* scpart   = logZpart + 64;

    crf_prep<<<16, 256, 0, stream>>>(trans, EThi, ETlo);
    crf_forward<<<B_ / 16, 128, 0, stream>>>(emit, mask, strans, etrans, trans,
                                             target, EThi, ETlo, logZpart, scpart);
    crf_final<<<1, 1, 0, stream>>>(logZpart, scpart, (float*)d_out);
}
